// Retention_55714315763737
// MI455X (gfx1250) — matrix operand order repaired and checked by reading, not compiled, not run on hardware
//
#include <hip/hip_runtime.h>
#include <hip/hip_bf16.h>

// ---------------------------------------------------------------------------
// Retention: out = ((q @ k^T)/sqrt(d) * alpha^(i-j) causal) @ v
// K1: qkv GEMM -> workspace:  qk [32768,1536] bf16 (q pre-scaled by 1/sqrt(d)),
//                             vT [32,768,1024] bf16 (V stored transposed)
// K2: per (batch, 64-row block): S = Q K^T (WMMA bf16, decay in f32 regs) ->
//     bf16 LDS slab; then O = S V from vT. Causality bounds both k-loops at
//     qr0+64 (scalar bound -> EXEC stays all-ones), halving matrix work.
// ---------------------------------------------------------------------------

typedef __attribute__((ext_vector_type(16))) __bf16 v16bf;
typedef __attribute__((ext_vector_type(2)))  __bf16 v2bf;
typedef __attribute__((ext_vector_type(8)))  float  v8f;
typedef __attribute__((ext_vector_type(4)))  unsigned int u32x4;
typedef __attribute__((ext_vector_type(4)))  int i32x4;
typedef __attribute__((address_space(1))) i32x4 i32x4_g;   // global (AS1)
typedef __attribute__((address_space(3))) i32x4 i32x4_l;   // LDS (AS3)

#define SEQ    1024
#define DIM    768
#define NDIM   2304                       // 3*768 (GEMM N)
#define QKLD   1536                       // row stride of qk slab (halves)
#define QSCALE 0.03608439182435161f       // 1/sqrt(768)
#define L2A    (-0.014499569695115089f)   // log2(0.99)

#define VT_BASE_HALVES ((size_t)32768 * QKLD)   // vT follows qk in workspace

#if defined(__has_builtin)
#if __has_builtin(__builtin_amdgcn_global_load_async_to_lds_b128) && \
    __has_builtin(__builtin_amdgcn_s_wait_asynccnt)
#define HAVE_ASYNC_LDS 1
#endif
#if __has_builtin(__builtin_amdgcn_cvt_pk_bf16_f32)
#define HAVE_CVT_PK 1
#endif
#endif
#ifndef HAVE_ASYNC_LDS
#define HAVE_ASYNC_LDS 0
#endif
#ifndef HAVE_CVT_PK
#define HAVE_CVT_PK 0
#endif

__device__ __forceinline__ unsigned short f2bf(float f) {
  unsigned int x = __float_as_uint(f);
  unsigned int r = x + 0x7FFFu + ((x >> 16) & 1u);   // round-to-nearest-even
  return (unsigned short)(r >> 16);
}

// two f32 -> packed 2xbf16 (one VALU op when v_cvt_pk_bf16_f32 is available)
__device__ __forceinline__ unsigned int pk2bf(float a, float b) {
#if HAVE_CVT_PK
  union { v2bf v; unsigned int u; } t;
  t.v = __builtin_amdgcn_cvt_pk_bf16_f32(a, b);
  return t.u;
#else
  return (unsigned int)f2bf(a) | ((unsigned int)f2bf(b) << 16);
#endif
}

__device__ __forceinline__ v16bf load_frag2(const unsigned short* p0,
                                            const unsigned short* p1) {
  union { v16bf v; u32x4 q[2]; } f;
  f.q[0] = *(const u32x4*)p0;   // 16B chunk (8 bf16)
  f.q[1] = *(const u32x4*)p1;   // 16B chunk (8 bf16)
  return f.v;
}

__device__ __forceinline__ v8f wmma_bf16(v16bf a, v16bf b, v8f c) {
  return __builtin_amdgcn_wmma_f32_16x16x32_bf16(false, a, false, b,
                                                 (short)0, c, false, false);
}

// 16B global -> LDS copy; async-DMA path (ASYNCcnt) on gfx1250.
__device__ __forceinline__ void cp16_g2l(const unsigned short* g,
                                         unsigned short* l) {
#if HAVE_ASYNC_LDS
  __builtin_amdgcn_global_load_async_to_lds_b128((i32x4_g*)g, (i32x4_l*)l,
                                                 0, 0);
#else
  *(u32x4*)l = *(const u32x4*)g;
#endif
}

__device__ __forceinline__ void cp_fence_lds() {
#if HAVE_ASYNC_LDS
  __builtin_amdgcn_s_wait_asynccnt(0);
#endif
  __syncthreads();
}

// ---------------------------------------------------------------------------
// Kernel 1: [32768,2304] = x[32768,768] @ W[768,2304] + b  -> bf16 qk + vT
// Workgroup: 256 thr (8 waves), tile 128x128; wave tile 64x32 (4x2 WMMA tiles)
// ---------------------------------------------------------------------------
__global__ __launch_bounds__(256) void qkv_gemm_kernel(
    const float* __restrict__ x, const float* __restrict__ W,
    const float* __restrict__ bias, unsigned short* __restrict__ qk,
    unsigned short* __restrict__ vT) {
  __shared__ unsigned short ldsA[128 * 40];   // [row][k] bf16, pad 32->40
  __shared__ unsigned short ldsB[128 * 40];   // [col][k] bf16 (W transposed)

  const int tid  = threadIdx.x;
  const int lane = tid & 31;
  const int wave = tid >> 5;
  const int waveM = wave >> 2;           // 0..1 -> 64-row half
  const int waveN = wave & 3;            // 0..3 -> 32-col quarter
  const int m0 = blockIdx.y * 128;
  const int n0 = blockIdx.x * 128;

  const int lrow = lane & 15;
  const int hi   = lane >> 4;            // lane half
  const int ab   = hi ? 16 : 0;          // A fragment K base
  const int bb   = hi ? 16 : 0;          // B fragment K base

  // fixed staging coordinates for this thread
  const int ar = tid >> 3, ak4 = tid & 7;     // A: 32 rows per pass
  const int br = tid >> 5, bc4 = tid & 31;    // B: 8 rows per pass

  v8f acc[4][2] = {};

  for (int kk = 0; kk < DIM; kk += 32) {
    // Issue global loads for this k-slice before the barrier (latency overlap)
    float4 va[4], vb[4];
#pragma unroll
    for (int it = 0; it < 4; ++it)
      va[it] = *(const float4*)(x + (size_t)(m0 + ar + it * 32) * DIM + kk +
                                ak4 * 4);
#pragma unroll
    for (int it = 0; it < 4; ++it)
      vb[it] = *(const float4*)(W + (size_t)(kk + br + it * 8) * NDIM + n0 +
                                bc4 * 4);
    __syncthreads();   // previous iteration's fragment reads done
#pragma unroll
    for (int it = 0; it < 4; ++it) {   // A tile: packed bf16 stores
      unsigned int* p = (unsigned int*)&ldsA[(ar + it * 32) * 40 + ak4 * 4];
      p[0] = pk2bf(va[it].x, va[it].y);
      p[1] = pk2bf(va[it].z, va[it].w);
    }
#pragma unroll
    for (int it = 0; it < 4; ++it) {   // B tile: transposed scatter
      const int r = br + it * 8;
      ldsB[(bc4 * 4 + 0) * 40 + r] = f2bf(vb[it].x);
      ldsB[(bc4 * 4 + 1) * 40 + r] = f2bf(vb[it].y);
      ldsB[(bc4 * 4 + 2) * 40 + r] = f2bf(vb[it].z);
      ldsB[(bc4 * 4 + 3) * 40 + r] = f2bf(vb[it].w);
    }
    __syncthreads();

    v16bf a[4], b[2];
#pragma unroll
    for (int i = 0; i < 4; ++i) {
      const unsigned short* p = &ldsA[(waveM * 64 + i * 16 + lrow) * 40 + ab];
      a[i] = load_frag2(p, p + 8);
    }
#pragma unroll
    for (int j = 0; j < 2; ++j) {
      const unsigned short* p = &ldsB[(waveN * 32 + j * 16 + lrow) * 40 + bb];
      b[j] = load_frag2(p, p + 8);
    }
#pragma unroll
    for (int i = 0; i < 4; ++i)
#pragma unroll
      for (int j = 0; j < 2; ++j)
        acc[i][j] = wmma_bf16(a[i], b[j], acc[i][j]);
  }

  // Epilogue. q/k/v split is uniform per workgroup: blockIdx.x<6 -> q,
  // <12 -> k, else v (boundaries are multiples of the 128-col tile).
  const int mh = hi * 8;
  if (blockIdx.x < 12) {                       // ---- q | k -> qk slab ----
    const float sc = (blockIdx.x < 6) ? QSCALE : 1.0f;
#pragma unroll
    for (int i = 0; i < 4; ++i)
#pragma unroll
      for (int j = 0; j < 2; ++j) {
        const int col = n0 + waveN * 32 + j * 16 + lrow;
        const float bv = bias[col];
#pragma unroll
        for (int r = 0; r < 8; ++r) {
          const int row = m0 + waveM * 64 + i * 16 + mh + r;
          qk[(size_t)row * QKLD + col] = f2bf((acc[i][j][r] + bv) * sc);
        }
      }
  } else {                                     // ---- v -> vT (transposed) ----
    const int bt = m0 >> 10;                   // whole block is one batch
    const int s0 = m0 & (SEQ - 1);
#pragma unroll
    for (int i = 0; i < 4; ++i)
#pragma unroll
      for (int j = 0; j < 2; ++j) {
        const int col = n0 + waveN * 32 + j * 16 + lrow;
        const float bv = bias[col];
        const int d = col - 2 * DIM;
#pragma unroll
        for (int r = 0; r < 8; ++r) {
          const int s = s0 + waveM * 64 + i * 16 + mh + r;
          vT[((size_t)bt * DIM + d) * SEQ + s] = f2bf(acc[i][j][r] + bv);
        }
      }
  }
}

// ---------------------------------------------------------------------------
// Kernel 2: per (batch, 64-query-row block):
//   Phase 1: S[64, 0..qr0+63] = Qblk K^T (decay in f32) -> bf16 LDS
//   Phase 2: O[64,768] = S V, kk < qr0+64 only (causality)
// LDS: S 64x1032 (129KB) + Q 64x776 (97KB)  (320KB WGP budget)
// ---------------------------------------------------------------------------
#define LDS_S_STRIDE 1032
#define LDS_Q_STRIDE 776
#define LDS_S_HALVES (64 * LDS_S_STRIDE)
#define LDS_Q_HALVES (64 * LDS_Q_STRIDE)
#define RET_LDS_BYTES ((LDS_S_HALVES + LDS_Q_HALVES) * 2)

__global__ __launch_bounds__(256) void retention_kernel(
    const unsigned short* __restrict__ qk, const unsigned short* __restrict__ vT,
    float* __restrict__ out) {
  extern __shared__ unsigned short lds[];
  unsigned short* ldsS = lds;                     // scores slab
  unsigned short* ldsQ = lds + LDS_S_HALVES;      // Q block

  const int tid  = threadIdx.x;
  const int lane = tid & 31;
  const int wave = tid >> 5;
  const int b    = blockIdx.y;
  const int qr0  = blockIdx.x * 64;
  const int kend = qr0 + 64;                  // causal bound (scalar)
  const size_t rowbase = (size_t)b * SEQ;

  const int lrow = lane & 15;
  const int hi   = lane >> 4;
  const int ab   = hi ? 16 : 0;
  const int bb   = hi ? 16 : 0;
  const int mh   = hi * 8;

  // Stage Q block (async DMA to LDS; bf16, no conversion needed)
  {
    const int qrow = tid >> 2;                // 0..63, 4 threads per row
    const int cb   = tid & 3;
    const unsigned short* gQ = qk + (rowbase + qr0 + qrow) * QKLD;
    unsigned short* lQ = ldsQ + qrow * LDS_Q_STRIDE;
#pragma unroll
    for (int it = 0; it < 24; ++it) {
      const int c = (cb + (it << 2)) << 3;    // uint4 column in halves
      cp16_g2l(gQ + c, lQ + c);
    }
  }
  cp_fence_lds();

  // ---- Phase 1: scores with decay (tiles above diagonal skipped) ----
  for (int ct = wave; ct * 16 < kend; ct += 8) {   // 16-wide key tiles
    const int n0 = ct * 16;
    const unsigned short* krow = qk + (rowbase + n0 + lrow) * QKLD + DIM + bb;
    v8f acc[4] = {};
    for (int kk = 0; kk < DIM; kk += 32) {
      v16bf a[4];
#pragma unroll
      for (int i = 0; i < 4; ++i) {
        const unsigned short* p =
            &ldsQ[(i * 16 + lrow) * LDS_Q_STRIDE + kk + ab];
        a[i] = load_frag2(p, p + 8);
      }
      // B = K^T fragment: contiguous 32B straight from global K row
      v16bf bf = load_frag2(krow + kk, krow + kk + 8);
      __builtin_prefetch((const void*)(krow + kk + 32), 0, 1);
#pragma unroll
      for (int i = 0; i < 4; ++i) acc[i] = wmma_bf16(a[i], bf, acc[i]);
    }
    // decay * causal, write bf16 scores to LDS
#pragma unroll
    for (int i = 0; i < 4; ++i)
#pragma unroll
      for (int r = 0; r < 8; ++r) {
        const int li = i * 16 + mh + r;
        const int dd = (qr0 + li) - (n0 + lrow);
        const float f = (dd >= 0) ? exp2f((float)dd * L2A) : 0.0f;
        ldsS[li * LDS_S_STRIDE + n0 + lrow] = f2bf(acc[i][r] * f);
      }
  }
  __syncthreads();

  // ---- Phase 2: O = S @ V, kk bounded by causality ----
  const int wc0 = wave * 96;                  // 96 output cols per wave
  v8f oacc[4][6] = {};
  const unsigned short* vbase = vT + ((size_t)b * DIM + wc0 + lrow) * SEQ + bb;
  for (int kk = 0; kk < kend; kk += 32) {
    v16bf a[4];
#pragma unroll
    for (int i = 0; i < 4; ++i) {
      const unsigned short* p =
          &ldsS[(i * 16 + lrow) * LDS_S_STRIDE + kk + ab];
      a[i] = load_frag2(p, p + 8);
    }
#pragma unroll
    for (int j = 0; j < 6; ++j) {
      const unsigned short* vp = vbase + (size_t)(j * 16) * SEQ + kk;
      v16bf bf = load_frag2(vp, vp + 8);
#pragma unroll
      for (int i = 0; i < 4; ++i) oacc[i][j] = wmma_bf16(a[i], bf, oacc[i][j]);
    }
  }

  // Store O (f32)
#pragma unroll
  for (int i = 0; i < 4; ++i)
#pragma unroll
    for (int j = 0; j < 6; ++j)
#pragma unroll
      for (int r = 0; r < 8; ++r) {
        const int row = qr0 + i * 16 + mh + r;
        const int col = wc0 + j * 16 + lrow;
        out[(rowbase + row) * DIM + col] = oacc[i][j][r];
      }
}

// ---------------------------------------------------------------------------
extern "C" void kernel_launch(void* const* d_in, const int* in_sizes, int n_in,
                              void* d_out, int out_size, void* d_ws, size_t ws_size,
                              hipStream_t stream) {
  const float* x    = (const float*)d_in[0];   // [32,1024,768]
  const float* Wqkv = (const float*)d_in[1];   // [768,2304]
  const float* bias = (const float*)d_in[2];   // [2304]
  float* out = (float*)d_out;
  unsigned short* qk = (unsigned short*)d_ws;              // [32768,1536] bf16
  unsigned short* vT = qk + VT_BASE_HALVES;                // [32,768,1024] bf16

  dim3 g1(NDIM / 128, (32 * SEQ) / 128);       // 18 x 256
  qkv_gemm_kernel<<<g1, 256, 0, stream>>>(x, Wqkv, bias, qk, vT);

  dim3 g2(SEQ / 64, 32);                       // 16 row-blocks x 32 batches
  retention_kernel<<<g2, 256, RET_LDS_BYTES, stream>>>(qk, vT, out);
}
